// Attention_64888365908310
// MI455X (gfx1250) — matrix operand order repaired and checked by reading, not compiled, not run on hardware
//
#include <hip/hip_runtime.h>

typedef __attribute__((ext_vector_type(16))) __bf16 v16bf;
typedef __attribute__((ext_vector_type(8)))  float  v8f;
typedef __attribute__((ext_vector_type(4)))  unsigned int u32x4;
typedef __attribute__((ext_vector_type(4)))  int i32x4;

#define GLD  72  // attention LDS row stride (ushorts): 144B, 16B-aligned
#define GLG  40  // gemm LDS row stride (ushorts): 80B, 16B-aligned

#if defined(__gfx1250__) && __has_builtin(__builtin_amdgcn_global_load_async_to_lds_b128)
#define HAVE_ASYNC 1
#else
#define HAVE_ASYNC 0
#endif

// 16B memory -> LDS copy. Async DMA (ASYNCcnt) on gfx1250, sync fallback otherwise.
__device__ __forceinline__ void cp_b128(const unsigned short* g, unsigned short* l) {
#if HAVE_ASYNC
  __builtin_amdgcn_global_load_async_to_lds_b128(
      (__attribute__((address_space(1))) i32x4*)g,
      (__attribute__((address_space(3))) i32x4*)l, 0, 0);
#else
  *(u32x4*)l = *(const u32x4*)g;
#endif
}

__device__ __forceinline__ void wait_cp() {
#if HAVE_ASYNC
#if __has_builtin(__builtin_amdgcn_s_wait_asynccnt)
  __builtin_amdgcn_s_wait_asynccnt(0);
#else
  asm volatile("s_wait_asynccnt 0x0" ::: "memory");
#endif
#endif
}

__device__ __forceinline__ unsigned short f2bf(float f) {
  union { float f; unsigned int u; } v; v.f = f;
  unsigned int r = v.u + 0x7fffu + ((v.u >> 16) & 1u);
  return (unsigned short)(r >> 16);
}

union FragBF { u32x4 q[2]; v16bf v; };

// A fragment (16x32 MxK bf16): lanes 0-15 row m hold K{0..7,16..23}; lanes 16-31 K{8..15,24..31}.
__device__ __forceinline__ v16bf frag_a_ld(const unsigned short* base, int ld) {
  int lane = threadIdx.x & 31;
  int m = lane & 15, hh = lane >> 4;
  const unsigned short* p = base + m * ld + hh * 8;
  FragBF f;
  f.q[0] = *(const u32x4*)(p);
  f.q[1] = *(const u32x4*)(p + 16);
  return f.v;
}

// B fragment (32x16 KxN bf16) from a TRANSPOSED tile (rows=N, cols=K):
// lanes 0-15 col n hold K{0..7,16..23}; lanes 16-31 K{8..15,24..31}; col n=lane&15.
__device__ __forceinline__ v16bf frag_bt_ld(const unsigned short* base, int ld) {
  int lane = threadIdx.x & 31;
  int n = lane & 15, hh = lane >> 4;
  const unsigned short* p = base + n * ld + hh * 8;
  FragBF f;
  f.q[0] = *(const u32x4*)(p);
  f.q[1] = *(const u32x4*)(p + 16);
  return f.v;
}

#define WMMA_BF16(a, b, c) \
  __builtin_amdgcn_wmma_f32_16x16x32_bf16(false, (a), false, (b), (short)0, (c), false, false)

// ---------------------------------------------------------------- prep kernels
__global__ __launch_bounds__(256) void k_convert(const float* __restrict__ in,
                                                 unsigned short* __restrict__ out, int n) {
  for (int i = blockIdx.x * 256 + threadIdx.x; i < n; i += gridDim.x * 256)
    out[i] = f2bf(in[i]);
}

// out[n*rows + k] = bf16(in[k*cols + n])
__global__ __launch_bounds__(256) void k_transpose(const float* __restrict__ in,
                                                   unsigned short* __restrict__ out,
                                                   int rows, int cols) {
  int total = rows * cols;
  for (int i = blockIdx.x * 256 + threadIdx.x; i < total; i += gridDim.x * 256) {
    int n = i / rows, k = i - n * rows;
    out[i] = f2bf(in[(size_t)k * cols + n]);
  }
}

// ---------------------------------------------------------------- QKV GEMM
// C[8192,3072] = Xbf[8192,1024] @ Wqkv; epilogue scatters q/k (row-major) and v (transposed).
__global__ __launch_bounds__(256) void k_gemm_qkv(const unsigned short* __restrict__ A,
                                                  const unsigned short* __restrict__ Bt,
                                                  const float* __restrict__ bias,
                                                  unsigned short* __restrict__ q_out,
                                                  unsigned short* __restrict__ k_out,
                                                  unsigned short* __restrict__ v_out) {
  __shared__ __attribute__((aligned(16))) unsigned short sA[2][128 * GLG];
  __shared__ __attribute__((aligned(16))) unsigned short sB[2][128 * GLG];
  const int tid = threadIdx.x;
  const int m0 = blockIdx.y * 128, n0 = blockIdx.x * 128;
  const int w = tid >> 5, wm = (w >> 2) * 64, wn = (w & 3) * 32;
  const int lrow = tid >> 1, lhalf = tid & 1;  // each thread stages 16 ushorts per tile

  const unsigned short* gA = A + (size_t)(m0 + lrow) * 1024 + lhalf * 16;
  const unsigned short* gB = Bt + (size_t)(n0 + lrow) * 1024 + lhalf * 16;
  const int lofs = lrow * GLG + lhalf * 16;

  v8f acc[4][2];
#pragma unroll
  for (int i = 0; i < 4; i++)
#pragma unroll
    for (int j = 0; j < 2; j++)
#pragma unroll
      for (int e = 0; e < 8; e++) acc[i][j][e] = 0.0f;

  // prologue: stage k-block 0 into buffer 0
  cp_b128(gA, &sA[0][lofs]);     cp_b128(gA + 8, &sA[0][lofs + 8]);
  cp_b128(gB, &sB[0][lofs]);     cp_b128(gB + 8, &sB[0][lofs + 8]);
  wait_cp();
  __syncthreads();

  for (int kblk = 0; kblk < 1024; kblk += 32) {
    const int cur = (kblk >> 5) & 1;
    if (kblk + 32 < 1024) {  // stage next block while computing this one
      cp_b128(gA + kblk + 32, &sA[cur ^ 1][lofs]);
      cp_b128(gA + kblk + 40, &sA[cur ^ 1][lofs + 8]);
      cp_b128(gB + kblk + 32, &sB[cur ^ 1][lofs]);
      cp_b128(gB + kblk + 40, &sB[cur ^ 1][lofs + 8]);
    }
    v16bf b0 = frag_bt_ld(&sB[cur][(wn + 0) * GLG], GLG);
    v16bf b1 = frag_bt_ld(&sB[cur][(wn + 16) * GLG], GLG);
#pragma unroll
    for (int i = 0; i < 4; i++) {
      v16bf af = frag_a_ld(&sA[cur][(wm + i * 16) * GLG], GLG);
      acc[i][0] = WMMA_BF16(af, b0, acc[i][0]);
      acc[i][1] = WMMA_BF16(af, b1, acc[i][1]);
    }
    wait_cp();
    __syncthreads();
  }

  const int lane = tid & 31, cn = lane & 15, hh = lane >> 4;
#pragma unroll
  for (int j = 0; j < 2; j++) {
    int nglob = n0 + wn + j * 16 + cn;
    float bv = bias[nglob];
    int sec = nglob >> 10, dcol = nglob & 1023;
    int head = dcol >> 6, dh = dcol & 63;
#pragma unroll
    for (int i = 0; i < 4; i++) {
#pragma unroll
      for (int r = 0; r < 8; r++) {
        int mglob = m0 + wm + i * 16 + hh * 8 + r;
        int bidx = mglob >> 11, srow = mglob & 2047;
        unsigned short val = f2bf(acc[i][j][r] + bv);
        size_t bh = (size_t)(bidx * 16 + head);
        if (sec == 0)      q_out[(bh * 2048 + srow) * 64 + dh] = val;
        else if (sec == 1) k_out[(bh * 2048 + srow) * 64 + dh] = val;
        else               v_out[(bh * 64 + dh) * 2048 + srow] = val;
      }
    }
  }
}

// ---------------------------------------------------------------- flash attention
__global__ __launch_bounds__(256) void k_attn(const unsigned short* __restrict__ qb,
                                              const unsigned short* __restrict__ kbuf,
                                              const unsigned short* __restrict__ vtb,
                                              unsigned short* __restrict__ attnb) {
  __shared__ __attribute__((aligned(16))) unsigned short sQ[128 * GLD];  // reused per-wave as P
  __shared__ __attribute__((aligned(16))) unsigned short sK[2][64 * GLD];
  __shared__ __attribute__((aligned(16))) unsigned short sV[2][64 * GLD];
  const int tid = threadIdx.x;
  const int bh = blockIdx.y;                 // b*16 + head
  const int b = bh >> 4, head = bh & 15;
  const int q0 = blockIdx.x * 128;
  const size_t hbase = (size_t)bh * 2048 * 64;

  { // stage Q block [128 x 64] (async)
    int qrow = tid >> 1, qhalf = tid & 1;    // 32 ushorts per thread
    const unsigned short* g = qb + hbase + (size_t)(q0 + qrow) * 64 + qhalf * 32;
    unsigned short* l = sQ + qrow * GLD + qhalf * 32;
    cp_b128(g, l); cp_b128(g + 8, l + 8); cp_b128(g + 16, l + 16); cp_b128(g + 24, l + 24);
  }
  // K/V staging addresses (per thread: 2x16B per tile)
  const int kr = tid >> 2, kc = (tid & 3) * 16;
  const unsigned short* gK = kbuf + hbase + (size_t)kr * 64 + kc;    // + j*64
  const unsigned short* gV = vtb + hbase + (size_t)kr * 2048 + kc;   // + j
  const int kvofs = kr * GLD + kc;

  // prologue: stage key block 0 into buffer 0 (alongside Q)
  cp_b128(gK, &sK[0][kvofs]);  cp_b128(gK + 8, &sK[0][kvofs + 8]);
  cp_b128(gV, &sV[0][kvofs]);  cp_b128(gV + 8, &sV[0][kvofs + 8]);
  wait_cp();
  __syncthreads();

  const int w = tid >> 5;
  v16bf qa0 = frag_a_ld(sQ + (w * 16) * GLD + 0, GLD);
  v16bf qa1 = frag_a_ld(sQ + (w * 16) * GLD + 32, GLD);
  unsigned short* sP = sQ + w * 16 * GLD;    // wave-private staging (rows this wave owns)

  v8f oacc[4];
  float m_st[8], l_st[8];
#pragma unroll
  for (int t = 0; t < 4; t++)
#pragma unroll
    for (int e = 0; e < 8; e++) oacc[t][e] = 0.0f;
#pragma unroll
  for (int r = 0; r < 8; r++) { m_st[r] = -__builtin_inff(); l_st[r] = 0.0f; }

  const int lane = tid & 31, cn = lane & 15, hh = lane >> 4;
  const float L2E = 1.44269504f;

  for (int j = 0; j < 2048; j += 64) {
    const int cur = (j >> 6) & 1;
    if (j + 64 < 2048) {  // stage next key block during compute
      cp_b128(gK + (size_t)(j + 64) * 64, &sK[cur ^ 1][kvofs]);
      cp_b128(gK + (size_t)(j + 64) * 64 + 8, &sK[cur ^ 1][kvofs + 8]);
      cp_b128(gV + j + 64, &sV[cur ^ 1][kvofs]);
      cp_b128(gV + j + 72, &sV[cur ^ 1][kvofs + 8]);
    }

    v8f sc[4];
#pragma unroll
    for (int t = 0; t < 4; t++) {
#pragma unroll
      for (int e = 0; e < 8; e++) sc[t][e] = 0.0f;
      sc[t] = WMMA_BF16(qa0, frag_bt_ld(&sK[cur][(t * 16) * GLD + 0], GLD), sc[t]);
      sc[t] = WMMA_BF16(qa1, frag_bt_ld(&sK[cur][(t * 16) * GLD + 32], GLD), sc[t]);
    }
#pragma unroll
    for (int t = 0; t < 4; t++)
#pragma unroll
      for (int r = 0; r < 8; r++) sc[t][r] *= 0.125f;  // 1/sqrt(64)

#pragma unroll
    for (int r = 0; r < 8; r++) {  // online softmax (lanes 0-15: row r; 16-31: row r+8)
      float mx = fmaxf(fmaxf(sc[0][r], sc[1][r]), fmaxf(sc[2][r], sc[3][r]));
      mx = fmaxf(mx, __shfl_xor(mx, 1, 16));
      mx = fmaxf(mx, __shfl_xor(mx, 2, 16));
      mx = fmaxf(mx, __shfl_xor(mx, 4, 16));
      mx = fmaxf(mx, __shfl_xor(mx, 8, 16));
      float nm = fmaxf(m_st[r], mx);
      float alpha = exp2f((m_st[r] - nm) * L2E);
      m_st[r] = nm;
      float rs = 0.0f;
#pragma unroll
      for (int t = 0; t < 4; t++) {
        float p = exp2f((sc[t][r] - nm) * L2E);
        sc[t][r] = p; rs += p;
      }
      rs += __shfl_xor(rs, 1, 16);
      rs += __shfl_xor(rs, 2, 16);
      rs += __shfl_xor(rs, 4, 16);
      rs += __shfl_xor(rs, 8, 16);
      l_st[r] = l_st[r] * alpha + rs;
#pragma unroll
      for (int t = 0; t < 4; t++) oacc[t][r] *= alpha;
    }

    // C-layout -> A-layout via wave-private LDS (DS in-order within a wave)
#pragma unroll
    for (int t = 0; t < 4; t++)
#pragma unroll
      for (int r = 0; r < 8; r++)
        sP[(hh * 8 + r) * GLD + t * 16 + cn] = f2bf(sc[t][r]);

    v16bf pa0 = frag_a_ld(sP + 0, GLD);
    v16bf pa1 = frag_a_ld(sP + 32, GLD);
#pragma unroll
    for (int t = 0; t < 4; t++) {
      oacc[t] = WMMA_BF16(pa0, frag_bt_ld(&sV[cur][(t * 16) * GLD + 0], GLD), oacc[t]);
      oacc[t] = WMMA_BF16(pa1, frag_bt_ld(&sV[cur][(t * 16) * GLD + 32], GLD), oacc[t]);
    }

    wait_cp();
    __syncthreads();
  }

#pragma unroll
  for (int r = 0; r < 8; r++) {
    float inv = 1.0f / l_st[r];
    int srow = q0 + w * 16 + hh * 8 + r;
#pragma unroll
    for (int t = 0; t < 4; t++)
      attnb[((size_t)(b * 2048 + srow)) * 1024 + head * 64 + t * 16 + cn] =
          f2bf(oacc[t][r] * inv);
  }
}

// ---------------------------------------------------------------- projection GEMM
__global__ __launch_bounds__(256) void k_gemm_proj(const unsigned short* __restrict__ A,
                                                   const unsigned short* __restrict__ Bt,
                                                   const float* __restrict__ bias,
                                                   float* __restrict__ out) {
  __shared__ __attribute__((aligned(16))) unsigned short sA[2][128 * GLG];
  __shared__ __attribute__((aligned(16))) unsigned short sB[2][128 * GLG];
  const int tid = threadIdx.x;
  const int m0 = blockIdx.y * 128, n0 = blockIdx.x * 128;
  const int w = tid >> 5, wm = (w >> 2) * 64, wn = (w & 3) * 32;
  const int lrow = tid >> 1, lhalf = tid & 1;

  const unsigned short* gA = A + (size_t)(m0 + lrow) * 1024 + lhalf * 16;
  const unsigned short* gB = Bt + (size_t)(n0 + lrow) * 1024 + lhalf * 16;
  const int lofs = lrow * GLG + lhalf * 16;

  v8f acc[4][2];
#pragma unroll
  for (int i = 0; i < 4; i++)
#pragma unroll
    for (int j = 0; j < 2; j++)
#pragma unroll
      for (int e = 0; e < 8; e++) acc[i][j][e] = 0.0f;

  cp_b128(gA, &sA[0][lofs]);     cp_b128(gA + 8, &sA[0][lofs + 8]);
  cp_b128(gB, &sB[0][lofs]);     cp_b128(gB + 8, &sB[0][lofs + 8]);
  wait_cp();
  __syncthreads();

  for (int kblk = 0; kblk < 1024; kblk += 32) {
    const int cur = (kblk >> 5) & 1;
    if (kblk + 32 < 1024) {
      cp_b128(gA + kblk + 32, &sA[cur ^ 1][lofs]);
      cp_b128(gA + kblk + 40, &sA[cur ^ 1][lofs + 8]);
      cp_b128(gB + kblk + 32, &sB[cur ^ 1][lofs]);
      cp_b128(gB + kblk + 40, &sB[cur ^ 1][lofs + 8]);
    }
    v16bf b0 = frag_bt_ld(&sB[cur][(wn + 0) * GLG], GLG);
    v16bf b1 = frag_bt_ld(&sB[cur][(wn + 16) * GLG], GLG);
#pragma unroll
    for (int i = 0; i < 4; i++) {
      v16bf af = frag_a_ld(&sA[cur][(wm + i * 16) * GLG], GLG);
      acc[i][0] = WMMA_BF16(af, b0, acc[i][0]);
      acc[i][1] = WMMA_BF16(af, b1, acc[i][1]);
    }
    wait_cp();
    __syncthreads();
  }

  const int lane = tid & 31, cn = lane & 15, hh = lane >> 4;
#pragma unroll
  for (int j = 0; j < 2; j++) {
    int nglob = n0 + wn + j * 16 + cn;
    float bv = bias[nglob];
#pragma unroll
    for (int i = 0; i < 4; i++)
#pragma unroll
      for (int r = 0; r < 8; r++) {
        int mglob = m0 + wm + i * 16 + hh * 8 + r;
        out[(size_t)mglob * 1024 + nglob] = acc[i][j][r] + bv;
      }
  }
}

// ---------------------------------------------------------------- launch
extern "C" void kernel_launch(void* const* d_in, const int* in_sizes, int n_in,
                              void* d_out, int out_size, void* d_ws, size_t ws_size,
                              hipStream_t stream) {
  const float* x      = (const float*)d_in[0];
  const float* w_qkv  = (const float*)d_in[1];
  const float* b_qkv  = (const float*)d_in[2];
  const float* w_proj = (const float*)d_in[3];
  const float* b_proj = (const float*)d_in[4];
  float* out = (float*)d_out;

  const size_t XN = 8192ull * 1024;        // x bf16
  const size_t WQ = 3072ull * 1024;        // w_qkv^T bf16
  const size_t WP = 1024ull * 1024;        // w_proj^T bf16
  const size_t HB = 4ull * 16 * 2048 * 64; // per q/k/vT buffer

  unsigned short* ws     = (unsigned short*)d_ws;
  unsigned short* xbf    = ws;
  unsigned short* wqkvT  = xbf + XN;
  unsigned short* wprojT = wqkvT + WQ;
  unsigned short* qbuf   = wprojT + WP;
  unsigned short* kbuf   = qbuf + HB;
  unsigned short* vtbuf  = kbuf + HB;
  unsigned short* attnb  = vtbuf + HB;

  k_convert<<<2048, 256, 0, stream>>>(x, xbf, (int)XN);
  k_transpose<<<2048, 256, 0, stream>>>(w_qkv, wqkvT, 1024, 3072);
  k_transpose<<<1024, 256, 0, stream>>>(w_proj, wprojT, 1024, 1024);
  k_gemm_qkv<<<dim3(24, 64), 256, 0, stream>>>(xbf, wqkvT, b_qkv, qbuf, kbuf, vtbuf);
  k_attn<<<dim3(16, 64), 256, 0, stream>>>(qbuf, kbuf, vtbuf, attnb);
  k_gemm_proj<<<dim3(8, 64), 256, 0, stream>>>(attnb, wprojT, b_proj, out);
}
